// ManifoldAttention_25374666785218
// MI455X (gfx1250) — matrix operand order repaired and checked by reading, not compiled, not run on hardware
//
#include <hip/hip_runtime.h>

// ---------------------------------------------------------------------------
// ManifoldAttention for MI455X (gfx1250, wave32, WMMA bf16).
// B=4, N=1024, C=768, H=12, HD=64.
// cross = Q@K^T with Q=qq^T,K=kk^T is reduced to q(q^Tk)k^T (rank-64),
// qn/kn via 64x64 Grams -> total ~45 GFLOP, ~0.6 GB HBM traffic.
// Score kernel stages operand tiles via global_load_async_to_lds_b128
// (ASYNCcnt path) and feeds WMMA from LDS.
// ---------------------------------------------------------------------------

#define DEV __device__ __forceinline__

constexpr int Bn  = 4;
constexpr int Nn  = 1024;
constexpr int Cn  = 768;
constexpr int Hn  = 12;
constexpr int HDn = 64;
constexpr int C3n = 3 * Cn;          // 2304
constexpr float SCALE = 0.125f;      // HD^-0.5

typedef __attribute__((ext_vector_type(16))) __bf16 bf16x16;
typedef __attribute__((ext_vector_type(8)))  __bf16 bf16x8;
typedef __attribute__((ext_vector_type(8)))  float  f32x8;

DEV __bf16 f2bf(float f) {
  unsigned u = __builtin_bit_cast(unsigned, f);
  u += 0x7FFFu + ((u >> 16) & 1u);               // round-to-nearest-even
  unsigned short s = (unsigned short)(u >> 16);
  return __builtin_bit_cast(__bf16, s);
}
DEV float bf2f(__bf16 b) {
  unsigned u = ((unsigned)__builtin_bit_cast(unsigned short, b)) << 16;
  return __builtin_bit_cast(float, u);
}

DEV f32x8 wmma_bf16(bf16x16 a, bf16x16 b, f32x8 c) {
  // (neg_a, A, neg_b, B, c_mod, C, reuse_a, reuse_b)
  return __builtin_amdgcn_wmma_f32_16x16x32_bf16(false, a, false, b, (short)0, c,
                                                 false, false);
}

DEV bf16x16 ld16(const __bf16* p0, const __bf16* p1) {
  bf16x8 lo = *(const bf16x8*)p0;
  bf16x8 hi = *(const bf16x8*)p1;
  return __builtin_shufflevector(lo, hi, 0,1,2,3,4,5,6,7,8,9,10,11,12,13,14,15);
}
// A operand (16x32 bf16, M x K, row-major src, stride ld).
// lane L: row = L&15 ; elems 0..7 = K kb..kb+7, elems 8..15 = K kb+16..kb+23,
// kb = 8*(L>>4)   (ISA 05_wmma 16-bit A layout).
DEV bf16x16 load_a(const __bf16* base, int ld, int lane) {
  const int r = lane & 15, kb = (lane >> 4) * 8;
  const __bf16* p = base + (size_t)r * ld + kb;
  return ld16(p, p + 16);
}
// B operand (32x16 bf16, K x N) sourced from row-major W[n][k]:
// lane L: col n = L&15 ; elems 0..7 = K kb..kb+7, elems 8..15 = K kb+16..kb+23, kb = 8*(L>>4).
DEV bf16x16 load_b(const __bf16* base, int ld, int lane) {
  const int n = lane & 15, kb = (lane >> 4) * 8;
  const __bf16* p = base + (size_t)n * ld + kb;
  return ld16(p, p + 16);
}

// async global -> LDS copy of one 16-byte chunk (per-lane), ASYNCcnt-tracked.
DEV void async_b128(unsigned lds_addr, unsigned long long gaddr) {
  asm volatile("global_load_async_to_lds_b128 %0, %1, off"
               :: "v"(lds_addr), "v"(gaddr)
               : "memory");
}
DEV void wait_async0() {
  asm volatile("s_wait_asynccnt 0x0" ::: "memory");
}

// ---------------------------------------------------------------------------
// K0: convert weights f32 -> bf16 (qkv_w [2304,768], proj_w [768,768])
// ---------------------------------------------------------------------------
__global__ void k_convert_w(const float* __restrict__ qkv_w,
                            const float* __restrict__ proj_w,
                            __bf16* __restrict__ wB, __bf16* __restrict__ pB) {
  const int nq = C3n * Cn, np = Cn * Cn;
  int idx = blockIdx.x * 256 + threadIdx.x;
  if (idx < nq)            wB[idx] = f2bf(qkv_w[idx]);
  else if (idx < nq + np)  pB[idx - nq] = f2bf(proj_w[idx - nq]);
}

// ---------------------------------------------------------------------------
// K1: QKV GEMM  [B*N=4096, 768] x [2304, 768]^T, epilogue scatters to
// q,k [b,h,n,d] and v^T [b,h,d,n] in bf16.  128x128 tile / 8 waves.
// ---------------------------------------------------------------------------
__global__ void k_qkv_gemm(const float* __restrict__ x, const __bf16* __restrict__ wB,
                           __bf16* __restrict__ qB, __bf16* __restrict__ kB,
                           __bf16* __restrict__ vT) {
  __shared__ __attribute__((aligned(16))) __bf16 As[128][40];
  const int tid = threadIdx.x;
  const int wave = tid >> 5, lane = tid & 31;
  const int m0 = blockIdx.y * 128, n0 = blockIdx.x * 128;
  const int wm = (wave >> 2) * 64, wn = (wave & 3) * 32;
  f32x8 acc[4][2] = {};

  for (int k0 = 0; k0 < Cn; k0 += 32) {
    // stage x tile 128x32 -> bf16 LDS (each thread: half a row, 16 f32)
    const int r = tid >> 1, cbase = (tid & 1) * 16;
    const float* xp = x + (size_t)(m0 + r) * Cn + k0 + cbase;
    __builtin_prefetch(xp + 32, 0, 1);   // next K-step -> global_prefetch_b8
    #pragma unroll
    for (int e = 0; e < 16; e += 4) {
      float4 v4 = *(const float4*)(xp + e);
      As[r][cbase + e + 0] = f2bf(v4.x);
      As[r][cbase + e + 1] = f2bf(v4.y);
      As[r][cbase + e + 2] = f2bf(v4.z);
      As[r][cbase + e + 3] = f2bf(v4.w);
    }
    __syncthreads();
    bf16x16 afr[4];
    #pragma unroll
    for (int fi = 0; fi < 4; ++fi) afr[fi] = load_a(&As[wm + fi * 16][0], 40, lane);
    #pragma unroll
    for (int fj = 0; fj < 2; ++fj) {
      bf16x16 bfr = load_b(wB + (size_t)(n0 + wn + fj * 16) * Cn + k0, Cn, lane);
      #pragma unroll
      for (int fi = 0; fi < 4; ++fi) acc[fi][fj] = wmma_bf16(afr[fi], bfr, acc[fi][fj]);
    }
    __syncthreads();
  }

  const int r0 = (lane >> 4) * 8, nl = lane & 15;
  #pragma unroll
  for (int fi = 0; fi < 4; ++fi)
    #pragma unroll
    for (int fj = 0; fj < 2; ++fj) {
      const int gn = n0 + wn + fj * 16 + nl;
      const int sel = gn / Cn, rem = gn % Cn;
      const int hh = rem >> 6, d = rem & 63;
      #pragma unroll
      for (int e = 0; e < 8; ++e) {
        const int gm = m0 + wm + fi * 16 + r0 + e;
        const int bb = gm >> 10, irow = gm & 1023;
        const __bf16 bv = f2bf(acc[fi][fj][e]);
        if (sel == 0)      qB[((size_t)(bb * Hn + hh) * Nn + irow) * HDn + d] = bv;
        else if (sel == 1) kB[((size_t)(bb * Hn + hh) * Nn + irow) * HDn + d] = bv;
        else               vT[((size_t)(bb * Hn + hh) * HDn + d) * Nn + irow] = bv;
      }
    }
}

// ---------------------------------------------------------------------------
// K2: per (b,h): Grams Gq=q^Tq, Gk=k^Tk, M=q^Tk (64x64 in LDS), then
// qn_i = q_i Gq q_i^T, kn_j = k_j Gk k_j^T, q' = q @ M (bf16).
// ---------------------------------------------------------------------------
__global__ void k_gram(const __bf16* __restrict__ qB, const __bf16* __restrict__ kB,
                       float* __restrict__ qn, float* __restrict__ kn,
                       __bf16* __restrict__ qpB) {
  const int bh = blockIdx.x;       // 0..47
  const int t = threadIdx.x;       // 0..255
  __shared__ float sGq[64][64], sGk[64][64], sM[64][64];
  __shared__ __bf16 qrow[16][64], krow[16][64];
  const __bf16* qh = qB + (size_t)bh * Nn * HDn;
  const __bf16* kh = kB + (size_t)bh * Nn * HDn;

  const int d = t >> 2, e0 = (t & 3) * 16;
  float aq[16] = {}, ak[16] = {}, am[16] = {};
  for (int ic = 0; ic < Nn; ic += 16) {
    #pragma unroll
    for (int u = 0; u < 4; ++u) {
      const int idx = t * 4 + u, rr = idx >> 6, cc = idx & 63;
      qrow[rr][cc] = qh[(size_t)(ic + rr) * HDn + cc];
      krow[rr][cc] = kh[(size_t)(ic + rr) * HDn + cc];
    }
    __syncthreads();
    for (int ii = 0; ii < 16; ++ii) {
      const float qd = bf2f(qrow[ii][d]);
      const float kd = bf2f(krow[ii][d]);
      #pragma unroll
      for (int e = 0; e < 16; ++e) {
        const float qe = bf2f(qrow[ii][e0 + e]);
        const float ke = bf2f(krow[ii][e0 + e]);
        aq[e] += qd * qe;  ak[e] += kd * ke;  am[e] += qd * ke;
      }
    }
    __syncthreads();
  }
  #pragma unroll
  for (int e = 0; e < 16; ++e) {
    sGq[d][e0 + e] = aq[e];  sGk[d][e0 + e] = ak[e];  sM[d][e0 + e] = am[e];
  }
  __syncthreads();

  for (int rr = t; rr < Nn; rr += 256) {
    float qv[64], kv[64];
    #pragma unroll
    for (int dd = 0; dd < 64; ++dd) {
      qv[dd] = bf2f(qh[(size_t)rr * HDn + dd]);
      kv[dd] = bf2f(kh[(size_t)rr * HDn + dd]);
    }
    float qacc = 0.f, kacc = 0.f;
    for (int ee = 0; ee < 64; ++ee) {
      float t1 = 0.f, t2 = 0.f, t3 = 0.f;
      #pragma unroll
      for (int dd = 0; dd < 64; ++dd) {
        t1 += qv[dd] * sGq[dd][ee];
        t2 += kv[dd] * sGk[dd][ee];
        t3 += qv[dd] * sM[dd][ee];
      }
      qacc += t1 * qv[ee];
      kacc += t2 * kv[ee];
      qpB[((size_t)bh * Nn + rr) * HDn + ee] = f2bf(t3);
    }
    qn[(size_t)bh * Nn + rr] = qacc;
    kn[(size_t)bh * Nn + rr] = kacc;
  }
}

// ---------------------------------------------------------------------------
// K3 (per batch): channel scores.  S = scale * q k^T ; R = sqrt(max(qn+kn-2*q'k^T,0))
// -> chan[h][i][j], chan[12+h][i][j] (bf16).  128x128 tile, wave = 64x32.
// The three 128x64 bf16 operand tiles (q, q', k) are contiguous in memory and
// are staged once per workgroup via async global->LDS (ASYNCcnt), then all
// WMMA fragments come from LDS (removes 4x/2x cross-wave redundant reads).
// ---------------------------------------------------------------------------
__global__ void k_score(const __bf16* __restrict__ qB, const __bf16* __restrict__ qpB,
                        const __bf16* __restrict__ kB,
                        const float* __restrict__ qn, const float* __restrict__ kn,
                        __bf16* __restrict__ chan, int b) {
  const int h = blockIdx.z;
  const int i0 = blockIdx.y * 128, j0 = blockIdx.x * 128;
  const int wave = threadIdx.x >> 5, lane = threadIdx.x & 31;
  const int wm = (wave >> 2) * 64, wn = (wave & 3) * 32;
  const __bf16* qh  = qB  + (size_t)(b * Hn + h) * Nn * HDn;
  const __bf16* qph = qpB + (size_t)(b * Hn + h) * Nn * HDn;
  const __bf16* kh  = kB  + (size_t)(b * Hn + h) * Nn * HDn;

  __shared__ __attribute__((aligned(16))) __bf16 Qs[128][64];   // 16 KB
  __shared__ __attribute__((aligned(16))) __bf16 Ps[128][64];   // 16 KB
  __shared__ __attribute__((aligned(16))) __bf16 Ks[128][64];   // 16 KB

  {
    const unsigned long long gq = (unsigned long long)(uintptr_t)(qh  + (size_t)i0 * HDn);
    const unsigned long long gp = (unsigned long long)(uintptr_t)(qph + (size_t)i0 * HDn);
    const unsigned long long gk = (unsigned long long)(uintptr_t)(kh  + (size_t)j0 * HDn);
    const unsigned lq = (unsigned)(uintptr_t)&Qs[0][0];
    const unsigned lp = (unsigned)(uintptr_t)&Ps[0][0];
    const unsigned lk = (unsigned)(uintptr_t)&Ks[0][0];
    #pragma unroll
    for (int u = 0; u < 4; ++u) {
      const unsigned off = (threadIdx.x + u * 256) * 16u;   // 1024 x 16B chunks
      async_b128(lq + off, gq + off);
      async_b128(lp + off, gp + off);
      async_b128(lk + off, gk + off);
    }
    wait_async0();
    __syncthreads();
  }

  f32x8 accS[4][2] = {}, accC[4][2] = {};
  #pragma unroll
  for (int k0 = 0; k0 < HDn; k0 += 32) {
    bf16x16 bfr[2];
    #pragma unroll
    for (int fj = 0; fj < 2; ++fj)
      bfr[fj] = load_b(&Ks[wn + fj * 16][k0], HDn, lane);
    #pragma unroll
    for (int fi = 0; fi < 4; ++fi) {
      bf16x16 aS = load_a(&Qs[wm + fi * 16][k0], HDn, lane);
      bf16x16 aC = load_a(&Ps[wm + fi * 16][k0], HDn, lane);
      #pragma unroll
      for (int fj = 0; fj < 2; ++fj) {
        accS[fi][fj] = wmma_bf16(aS, bfr[fj], accS[fi][fj]);
        accC[fi][fj] = wmma_bf16(aC, bfr[fj], accC[fi][fj]);
      }
    }
  }

  const int r0 = (lane >> 4) * 8, nl = lane & 15;
  __bf16* chS = chan + (size_t)h * Nn * Nn;
  __bf16* chR = chan + (size_t)(Hn + h) * Nn * Nn;
  const float* qnh = qn + (size_t)(b * Hn + h) * Nn;
  const float* knh = kn + (size_t)(b * Hn + h) * Nn;
  #pragma unroll
  for (int fj = 0; fj < 2; ++fj) {
    const int gj = j0 + wn + fj * 16 + nl;
    const float knj = knh[gj];
    #pragma unroll
    for (int fi = 0; fi < 4; ++fi) {
      const int gib = i0 + wm + fi * 16 + r0;
      #pragma unroll
      for (int e = 0; e < 8; ++e) {
        const int gi = gib + e;
        chS[(size_t)gi * Nn + gj] = f2bf(accS[fi][fj][e] * SCALE);
        const float d2 = qnh[gi] + knj - 2.0f * accC[fi][fj][e];
        chR[(size_t)gi * Nn + gj] = f2bf(sqrtf(fmaxf(d2, 0.0f)));
      }
    }
  }
}

// ---------------------------------------------------------------------------
// K4 (per batch): 1x1 conv over 24 channels + row softmax -> A bf16 [12,N,N].
// One block per row i; fused[12][1024] lives in LDS.
// ---------------------------------------------------------------------------
__global__ void k_mix_softmax(const __bf16* __restrict__ chan,
                              const float* __restrict__ conv_w,
                              __bf16* __restrict__ Abuf) {
  const int i = blockIdx.x;
  const int t = threadIdx.x;
  __shared__ float fused[12][Nn];
  __shared__ float wc[12][24];
  __shared__ float red[256];

  for (int u = t; u < 12 * 24; u += 256) wc[u / 24][u % 24] = conv_w[u];
  __syncthreads();

  for (int j = t; j < Nn; j += 256) {
    float cv[24];
    #pragma unroll
    for (int c = 0; c < 24; ++c) cv[c] = bf2f(chan[(size_t)c * Nn * Nn + (size_t)i * Nn + j]);
    #pragma unroll
    for (int hh = 0; hh < 12; ++hh) {
      float f = 0.f;
      #pragma unroll
      for (int c = 0; c < 24; ++c) f += wc[hh][c] * cv[c];
      fused[hh][j] = f;
    }
  }
  __syncthreads();

  for (int hh = 0; hh < 12; ++hh) {
    float pm = -3.0e38f;
    for (int j = t; j < Nn; j += 256) pm = fmaxf(pm, fused[hh][j]);
    red[t] = pm;
    __syncthreads();
    for (int s = 128; s > 0; s >>= 1) {
      if (t < s) red[t] = fmaxf(red[t], red[t + s]);
      __syncthreads();
    }
    const float mx = red[0];
    __syncthreads();
    float ps = 0.f;
    for (int j = t; j < Nn; j += 256) {
      const float e = __expf(fused[hh][j] - mx);
      fused[hh][j] = e;
      ps += e;
    }
    red[t] = ps;
    __syncthreads();
    for (int s = 128; s > 0; s >>= 1) {
      if (t < s) red[t] += red[t + s];
      __syncthreads();
    }
    const float inv = 1.0f / red[0];
    __syncthreads();
    for (int j = t; j < Nn; j += 256)
      Abuf[((size_t)hh * Nn + i) * Nn + j] = f2bf(fused[hh][j] * inv);
  }
}

// ---------------------------------------------------------------------------
// K5 (per batch): out_h = A_h [N,N] @ V_h  (B operand from v^T [d][j]).
// Block: 128 i x 64 d, wave = 64 i x 16 d.
// ---------------------------------------------------------------------------
__global__ void k_pv(const __bf16* __restrict__ Abuf, const __bf16* __restrict__ vT,
                     __bf16* __restrict__ Obuf, int b) {
  const int h = blockIdx.y;
  const int i0 = blockIdx.x * 128;
  const int wave = threadIdx.x >> 5, lane = threadIdx.x & 31;
  const int wm = (wave >> 2) * 64, wd = (wave & 3) * 16;
  const __bf16* Ah = Abuf + (size_t)h * Nn * Nn;
  const __bf16* Vh = vT + (size_t)(b * Hn + h) * HDn * Nn;

  f32x8 acc[4] = {};
  for (int k0 = 0; k0 < Nn; k0 += 32) {
    bf16x16 bfr = load_b(Vh + (size_t)wd * Nn + k0, Nn, lane);
    #pragma unroll
    for (int fi = 0; fi < 4; ++fi) {
      bf16x16 afr = load_a(Ah + (size_t)(i0 + wm + fi * 16) * Nn + k0, Nn, lane);
      acc[fi] = wmma_bf16(afr, bfr, acc[fi]);
    }
  }
  const int r0 = (lane >> 4) * 8, nl = lane & 15;
  #pragma unroll
  for (int fi = 0; fi < 4; ++fi)
    #pragma unroll
    for (int e = 0; e < 8; ++e) {
      const int gi = i0 + wm + fi * 16 + r0 + e;
      Obuf[((size_t)b * Nn + gi) * Cn + h * HDn + wd + nl] = f2bf(acc[fi][e]);
    }
}

// ---------------------------------------------------------------------------
// K6: final projection out[4096,768] = O @ proj_w^T + bias (f32 out).
// ---------------------------------------------------------------------------
__global__ void k_proj(const __bf16* __restrict__ Obuf, const __bf16* __restrict__ pB,
                       const float* __restrict__ proj_b, float* __restrict__ out) {
  const int m0 = blockIdx.y * 128, n0 = blockIdx.x * 128;
  const int wave = threadIdx.x >> 5, lane = threadIdx.x & 31;
  const int wm = (wave >> 2) * 64, wn = (wave & 3) * 32;
  f32x8 acc[4][2] = {};
  for (int k0 = 0; k0 < Cn; k0 += 32) {
    bf16x16 bfr[2];
    #pragma unroll
    for (int fj = 0; fj < 2; ++fj)
      bfr[fj] = load_b(pB + (size_t)(n0 + wn + fj * 16) * Cn + k0, Cn, lane);
    #pragma unroll
    for (int fi = 0; fi < 4; ++fi) {
      bf16x16 afr = load_a(Obuf + (size_t)(m0 + wm + fi * 16) * Cn + k0, Cn, lane);
      #pragma unroll
      for (int fj = 0; fj < 2; ++fj) acc[fi][fj] = wmma_bf16(afr, bfr[fj], acc[fi][fj]);
    }
  }
  const int r0 = (lane >> 4) * 8, nl = lane & 15;
  #pragma unroll
  for (int fi = 0; fi < 4; ++fi)
    #pragma unroll
    for (int fj = 0; fj < 2; ++fj) {
      const int gn = n0 + wn + fj * 16 + nl;
      const float bias = proj_b[gn];
      #pragma unroll
      for (int e = 0; e < 8; ++e) {
        const int gm = m0 + wm + fi * 16 + r0 + e;
        out[(size_t)gm * Cn + gn] = acc[fi][fj][e] + bias;
      }
    }
}

// ---------------------------------------------------------------------------
extern "C" void kernel_launch(void* const* d_in, const int* in_sizes, int n_in,
                              void* d_out, int out_size, void* d_ws, size_t ws_size,
                              hipStream_t stream) {
  const float* x      = (const float*)d_in[0];
  const float* qkv_w  = (const float*)d_in[1];
  const float* proj_w = (const float*)d_in[2];
  const float* proj_b = (const float*)d_in[3];
  const float* conv_w = (const float*)d_in[4];
  float* out = (float*)d_out;

  char* p = (char*)d_ws;
  auto alloc = [&](size_t bytes) -> char* {
    char* r = p;
    p += (bytes + 255) & ~(size_t)255;
    return r;
  };
  const size_t qkv_elems = (size_t)Bn * Hn * Nn * HDn;   // 3,145,728
  __bf16* qB  = (__bf16*)alloc(qkv_elems * 2);
  __bf16* kB  = (__bf16*)alloc(qkv_elems * 2);
  __bf16* vT  = (__bf16*)alloc(qkv_elems * 2);
  __bf16* qpB = (__bf16*)alloc(qkv_elems * 2);
  float*  qn  = (float*)alloc((size_t)Bn * Hn * Nn * 4);
  float*  kn  = (float*)alloc((size_t)Bn * Hn * Nn * 4);
  __bf16* wB  = (__bf16*)alloc((size_t)C3n * Cn * 2);
  __bf16* pB  = (__bf16*)alloc((size_t)Cn * Cn * 2);
  __bf16* Obuf = (__bf16*)alloc((size_t)Bn * Nn * Cn * 2);
  __bf16* chan = (__bf16*)alloc((size_t)2 * Hn * Nn * Nn * 2);   // per-batch, reused
  __bf16* Abuf = (__bf16*)alloc((size_t)Hn * Nn * Nn * 2);       // per-batch, reused

  // weights -> bf16
  {
    const int total = C3n * Cn + Cn * Cn;
    k_convert_w<<<(total + 255) / 256, 256, 0, stream>>>(qkv_w, proj_w, wB, pB);
  }
  // QKV projection
  k_qkv_gemm<<<dim3(C3n / 128, (Bn * Nn) / 128), 256, 0, stream>>>(x, wB, qB, kB, vT);
  // Grams, qn/kn, q' = qM
  k_gram<<<Bn * Hn, 256, 0, stream>>>(qB, kB, qn, kn, qpB);

  for (int b = 0; b < Bn; ++b) {
    k_score<<<dim3(Nn / 128, Nn / 128, Hn), 256, 0, stream>>>(qB, qpB, kB, qn, kn, chan, b);
    k_mix_softmax<<<Nn, 256, 0, stream>>>(chan, conv_w, Abuf);
    k_pv<<<dim3(Nn / 128, Hn), 256, 0, stream>>>(Abuf, vT, Obuf, b);
  }
  // final projection + bias
  k_proj<<<dim3(Cn / 128, (Bn * Nn) / 128), 256, 0, stream>>>(Obuf, pB, proj_b, out);
}
